// GPNNet_75265006895418
// MI455X (gfx1250) — matrix operand order repaired and checked by reading, not compiled, not run on hardware
//
#include <hip/hip_runtime.h>
#include <hip/hip_bf16.h>
#include <math.h>

typedef __attribute__((ext_vector_type(16))) __bf16 v16bf;
typedef __attribute__((ext_vector_type(8)))  float  v8f;
typedef __attribute__((ext_vector_type(4)))  unsigned int v4u;
typedef __attribute__((ext_vector_type(8)))  int v8i;
typedef __attribute__((ext_vector_type(4)))  int v4i;

#define IN_DIM   256
#define HID      256
#define LAT      16
#define NCLS     16
#define RLAYERS  6
#define KSTEPS   10
#define TELEPORT 0.1f
#define BM       64
#define KTILE    32

#if __has_builtin(__builtin_amdgcn_tensor_load_to_lds) && __has_builtin(__builtin_amdgcn_s_wait_tensorcnt)
#define USE_TDM 1
#else
#define USE_TDM 0
#endif

// ---------------------------------------------------------------- small utils
__device__ __forceinline__ float softplusf(float x) {
  return (x > 20.0f) ? x : log1pf(expf(x));
}

// -------------------------------------------------------- init deg + counts
__global__ void init_kernel(float* __restrict__ deg, float* __restrict__ counts, int n) {
  int i = blockIdx.x * blockDim.x + threadIdx.x;
  if (i < n) deg[i] = 0.0f;
  if (i < NCLS) counts[i] = 0.0f;
}

// ------------------------------------------------------------- class counts
__global__ void count_kernel(const int* __restrict__ y, const unsigned char* __restrict__ mask,
                             float* __restrict__ counts, int n) {
  int i = blockIdx.x * blockDim.x + threadIdx.x;
  if (i >= n) return;
  if (mask[i]) atomicAdd(&counts[y[i]], 1.0f);
}

// --------------------------------------------- log prior + evidence constants
__global__ void prior_kernel(const float* __restrict__ counts, float* __restrict__ logpr) {
  __shared__ float tot;
  int c = threadIdx.x;
  if (c == 0) {
    float s = 0.0f;
    for (int i = 0; i < NCLS; ++i) s += counts[i];
    tot = s;
  }
  __syncthreads();
  if (c < NCLS) {
    const float PI = 3.14159265358979323846f;
    float cst = 0.5f * (float)LAT * logf(4.0f * PI) + logf((float)NCLS);
    logpr[c] = logf(counts[c] / tot) + cst;
  }
}

// -------------------------------------------------------------------- degree
__global__ void deg_kernel(const int* __restrict__ col, float* __restrict__ deg, int ne) {
  int e = blockIdx.x * blockDim.x + threadIdx.x;
  if (e < ne) atomicAdd(&deg[col[e]], 1.0f);
}

__global__ void dinv_kernel(float* __restrict__ deg, int n) {
  int i = blockIdx.x * blockDim.x + threadIdx.x;
  if (i < n) {
    float d = deg[i];
    deg[i] = (d > 0.0f) ? rsqrtf(d) : 0.0f;
  }
}

__global__ void edgew_kernel(const int* __restrict__ row, const int* __restrict__ col,
                             const float* __restrict__ dinv, float* __restrict__ wE, int ne) {
  int e = blockIdx.x * blockDim.x + threadIdx.x;
  if (e < ne) wE[e] = dinv[row[e]] * dinv[col[e]];
}

// ----------------------------------------- one-time W1^T / W2^T bf16 transpose
__global__ void prep_w_kernel(const float* __restrict__ W1, const float* __restrict__ W2,
                              __bf16* __restrict__ w1t, __bf16* __restrict__ w2t) {
  int i = blockIdx.x * blockDim.x + threadIdx.x;
  if (i < HID * IN_DIM) {
    int nn = i >> 8;            // output row (hidden unit)
    int kk = i & 255;           // input dim
    w1t[i] = (__bf16)W1[kk * HID + nn];
  }
  if (i < LAT * HID) {
    int nn = i >> 8;            // latent unit
    int kk = i & 255;           // hidden dim
    w2t[i] = (__bf16)W2[kk * LAT + nn];
  }
}

// ---------------------------------------------------------- fused WMMA encoder
// z = relu(x @ W1 + b1) @ W2 + b2    (per-block: 64 rows)
// W1^T K-slices are streamed into double-buffered LDS by the Tensor Data Mover,
// one TDM transfer in flight while the WMMA chain consumes the other buffer.
__global__ __launch_bounds__(256) void encoder_kernel(
    const float* __restrict__ x,  const __bf16* __restrict__ w1t, const float* __restrict__ b1,
    const __bf16* __restrict__ w2t, const float* __restrict__ b2, float* __restrict__ z, int n)
{
  __shared__ __bf16 lds_w1t[2][HID * KTILE];  // W1^T k-slices (double buffer) 2x16 KB
  __shared__ __bf16 lds_h[BM * HID];          // relu hidden tile              32 KB
  __shared__ __bf16 lds_w2t[LAT * HID];       // W2^T: [n][k]                   8 KB
  __shared__ float  lds_b1[HID];

  const int tid  = threadIdx.x;
  const int lane = tid & 31;
  const int wave = tid >> 5;                 // 0..7
  const int block_row = blockIdx.x * BM;
  const int ksel = (lane >> 4) * 16;         // A-fragment K base per half-wave

  // stage W2^T (bf16) and b1 once
  for (int i = tid; i < LAT * HID; i += 256) lds_w2t[i] = w2t[i];
  lds_b1[tid] = b1[tid];

#if USE_TDM
  // Tensor DMA descriptor (D#) for one 256-row x 32-col bf16 tile of W1^T.
  // Group1: data_size=2B; tensor_dim0=256, dim1=256; tile_dim0=32, tile_dim1=256;
  // tensor_dim0_stride=256. Group0 carries lds/global addresses per transfer.
  v8i g1;
  g1[0] = 0x00010000;            // data_size=2B (bits 17:16 = 1)
  g1[1] = (int)(256u << 16);     // tensor_dim0[15:0]=256 in bits 63:48
  g1[2] = (int)(256u << 16);     // tensor_dim1[15:0]=256 in bits 95:80
  g1[3] = (int)(32u  << 16);     // tile_dim0=32 in bits 127:112
  g1[4] = 256;                   // tile_dim1=256 in bits 143:128
  g1[5] = 256;                   // tensor_dim0_stride[31:0]=256 (bits 207:160)
  g1[6] = 0;
  g1[7] = 0;
  v4i gz4 = {};                  // groups 2/3 unused (2D tensor)
  v8i gz8 = {};                  // trailing descriptor payload (unused)

  auto issue_tdm = [&](int k0, int bufi) {
    unsigned long long gaddr = (unsigned long long)(const void*)(w1t + k0); // tile start
    v4u g0;
    g0.x = 1u;                                              // count=1, user mode
    g0.y = (unsigned)(unsigned long long)(void*)lds_w1t[bufi]; // LDS byte offset
    g0.z = (unsigned)(gaddr & 0xFFFFFFFFu);                 // global_addr[31:0]
    g0.w = (unsigned)((gaddr >> 32) & 0x01FFFFFFu) | (2u << 30); // addr[56:32] | type=2
    __builtin_amdgcn_tensor_load_to_lds(g0, g1, gz4, gz4, gz8, 0);
  };
  if (wave == 0) issue_tdm(0, 0);            // prologue: fill buffer 0
#endif

  // ---- stage 1: h = relu(x @ W1 + b1)
  const int mtile  = wave & 3;               // 4 M-tiles of 16 rows
  const int ntile0 = (wave >> 2) * 8;        // 8 N-tiles per wave

  v8f acc[8] = {};

  int arow = block_row + mtile * 16 + (lane & 15);
  if (arow >= n) arow = n - 1;               // clamp tail (stores guarded later)

  int it = 0;
  for (int k0 = 0; k0 < IN_DIM; k0 += KTILE, ++it) {
    const int cur = it & 1;
#if USE_TDM
    if (wave == 0) {
      if (k0 + KTILE < IN_DIM) {
        issue_tdm(k0 + KTILE, cur ^ 1);      // prefetch next slice into other buffer
        __builtin_amdgcn_s_wait_tensorcnt(1);// in-order: current slice complete
      } else {
        __builtin_amdgcn_s_wait_tensorcnt(0);
      }
    }
    __syncthreads();                         // current slice visible to all waves
#else
    __syncthreads();                         // previous slice fully consumed
    #pragma unroll 4
    for (int kk = 0; kk < KTILE; ++kk)
      lds_w1t[cur][tid * KTILE + kk] = w1t[tid * IN_DIM + k0 + kk];
    __syncthreads();
#endif

    // A fragment (16x32 bf16): lanes 0-15 hold K 0-15, lanes 16-31 K 16-31
    v16bf a;
    const float* xp = x + (size_t)arow * IN_DIM + k0 + ksel;
    #pragma unroll
    for (int i = 0; i < 8; ++i) a[i]     = (__bf16)xp[i];
    #pragma unroll
    for (int i = 0; i < 8; ++i) a[8 + i] = (__bf16)xp[8 + i];

    // preload all 8 B fragments, then run the WMMA chain back-to-back
    v16bf bf[8];
    #pragma unroll
    for (int t = 0; t < 8; ++t) {
      int nn = (ntile0 + t) * 16 + (lane & 15);
      const __bf16* bp = &lds_w1t[cur][nn * KTILE + (lane >> 4) * 16];
      #pragma unroll
      for (int i = 0; i < 16; ++i) bf[t][i] = bp[i];
    }
    #pragma unroll
    for (int t = 0; t < 8; ++t)
      acc[t] = __builtin_amdgcn_wmma_f32_16x16x32_bf16(
          false, a, false, bf[t], (short)0, acc[t], false, false);

    __syncthreads();                         // buffer consumed: safe for next TDM overwrite
  }

  // bias + relu, stash h tile to LDS as bf16 (C/D layout: VGPR g -> M=g(+8), lane -> N)
  #pragma unroll
  for (int t = 0; t < 8; ++t) {
    int ncol = (ntile0 + t) * 16 + (lane & 15);
    float bias = lds_b1[ncol];
    #pragma unroll
    for (int g = 0; g < 8; ++g) {
      int mrow = mtile * 16 + g + 8 * (lane >> 4);
      float v = acc[t][g] + bias;
      v = v > 0.0f ? v : 0.0f;
      lds_h[mrow * HID + ncol] = (__bf16)v;
    }
  }
  __syncthreads();

  // ---- stage 2: z = h @ W2 + b2  (4 waves, one 16x16 tile each)
  if (wave < 4) {
    v8f zacc = {};
    const int hrow = wave * 16 + (lane & 15);
    for (int k0 = 0; k0 < HID; k0 += KTILE) {
      v16bf a;
      const __bf16* hp = &lds_h[hrow * HID + k0 + ksel];
      #pragma unroll
      for (int i = 0; i < 8; ++i) a[i]     = hp[i];
      #pragma unroll
      for (int i = 0; i < 8; ++i) a[8 + i] = hp[8 + i];

      v16bf b;
      const __bf16* bp = &lds_w2t[(lane & 15) * HID + k0 + (lane >> 4) * 16];
      #pragma unroll
      for (int i = 0; i < 16; ++i) b[i] = bp[i];

      zacc = __builtin_amdgcn_wmma_f32_16x16x32_bf16(
          false, a, false, b, (short)0, zacc, false, false);
    }
    int ncol = lane & 15;
    float bias = b2[ncol];
    #pragma unroll
    for (int g = 0; g < 8; ++g) {
      int mrow = block_row + wave * 16 + g + 8 * (lane >> 4);
      if (mrow < n) z[(size_t)mrow * LAT + ncol] = zacc[g] + bias;
    }
  }
}

// --------------------------------------------- radial flow log-prob + evidence
__global__ __launch_bounds__(256) void flow_kernel(
    const float* __restrict__ z, const float* __restrict__ z0,
    const float* __restrict__ ap, const float* __restrict__ bpp,
    const float* __restrict__ logpr,
    float* __restrict__ beta_ft, float* __restrict__ prop0, int n)
{
  __shared__ float s_z0[NCLS * RLAYERS * LAT];
  __shared__ float s_ap[NCLS * RLAYERS];
  __shared__ float s_bp[NCLS * RLAYERS];
  __shared__ float s_lp[NCLS];

  int tid = threadIdx.x;
  for (int i = tid; i < NCLS * RLAYERS * LAT; i += 256) s_z0[i] = z0[i];
  for (int i = tid; i < NCLS * RLAYERS; i += 256) { s_ap[i] = ap[i]; s_bp[i] = bpp[i]; }
  if (tid < NCLS) s_lp[tid] = logpr[tid];
  __syncthreads();

  long long idx = (long long)blockIdx.x * blockDim.x + tid;
  if (idx >= (long long)n * NCLS) return;
  int node = (int)(idx >> 4);
  int c    = (int)(idx & 15);

  float zc[LAT];
  #pragma unroll
  for (int i = 0; i < LAT; ++i) zc[i] = z[(size_t)node * LAT + i];

  float logdet = 0.0f;
  #pragma unroll
  for (int l = 0; l < RLAYERS; ++l) {
    float alpha = softplusf(s_ap[c * RLAYERS + l]);
    float beta  = -alpha + softplusf(s_bp[c * RLAYERS + l]);
    const float* zz = &s_z0[(c * RLAYERS + l) * LAT];
    float diff[LAT], r2 = 0.0f;
    #pragma unroll
    for (int i = 0; i < LAT; ++i) { diff[i] = zc[i] - zz[i]; r2 += diff[i] * diff[i]; }
    float r  = sqrtf(r2);
    float h  = 1.0f / (alpha + r);
    float bh = beta * h;
    #pragma unroll
    for (int i = 0; i < LAT; ++i) zc[i] += bh * diff[i];
    logdet += (float)(LAT - 1) * log1pf(bh) + log1pf(bh - beta * r * h * h);
  }
  float s = 0.0f;
  #pragma unroll
  for (int i = 0; i < LAT; ++i) s += zc[i] * zc[i];
  const float PI = 3.14159265358979323846f;
  float logp = -0.5f * (s + (float)LAT * logf(2.0f * PI));
  float lq = logp + logdet + s_lp[c];
  lq = fminf(fmaxf(lq, -30.0f), 30.0f);
  float bft = expf(lq);
  beta_ft[idx] = bft;
  prop0[idx]   = bft;      // APPNP state starts at beta_ft
}

// ---------------------------------------------------------- APPNP iteration
__global__ void teleport_kernel(const float* __restrict__ beta_ft,
                                float* __restrict__ dst, long long total) {
  long long i = (long long)blockIdx.x * blockDim.x + threadIdx.x;
  if (i < total) dst[i] = TELEPORT * beta_ft[i];
}

__global__ void scatter_kernel(const int* __restrict__ row, const int* __restrict__ col,
                               const float* __restrict__ wE,
                               const float* __restrict__ src, float* __restrict__ dst, int ne) {
  int e = blockIdx.x * blockDim.x + threadIdx.x;
  if (e >= ne) return;
  int r = row[e], c = col[e];
  float wv = wE[e] * (1.0f - TELEPORT);
  const float4* sp = (const float4*)(src + (size_t)r * NCLS);
  float4 a0 = sp[0], a1 = sp[1], a2 = sp[2], a3 = sp[3];
  float* dp = dst + (size_t)c * NCLS;
  atomicAdd(dp + 0,  wv * a0.x); atomicAdd(dp + 1,  wv * a0.y);
  atomicAdd(dp + 2,  wv * a0.z); atomicAdd(dp + 3,  wv * a0.w);
  atomicAdd(dp + 4,  wv * a1.x); atomicAdd(dp + 5,  wv * a1.y);
  atomicAdd(dp + 6,  wv * a1.z); atomicAdd(dp + 7,  wv * a1.w);
  atomicAdd(dp + 8,  wv * a2.x); atomicAdd(dp + 9,  wv * a2.y);
  atomicAdd(dp + 10, wv * a2.z); atomicAdd(dp + 11, wv * a2.w);
  atomicAdd(dp + 12, wv * a3.x); atomicAdd(dp + 13, wv * a3.y);
  atomicAdd(dp + 14, wv * a3.z); atomicAdd(dp + 15, wv * a3.w);
}

__global__ void final_kernel(const float* __restrict__ src, float* __restrict__ out,
                             long long total) {
  long long i = (long long)blockIdx.x * blockDim.x + threadIdx.x;
  if (i < total) out[i] = 1.0f + src[i];
}

// ------------------------------------------------------------------- launcher
extern "C" void kernel_launch(void* const* d_in, const int* in_sizes, int n_in,
                              void* d_out, int out_size, void* d_ws, size_t ws_size,
                              hipStream_t stream) {
  const float* x   = (const float*)d_in[0];
  const float* W1  = (const float*)d_in[1];
  const float* b1  = (const float*)d_in[2];
  const float* W2  = (const float*)d_in[3];
  const float* b2  = (const float*)d_in[4];
  const float* z0  = (const float*)d_in[5];
  const float* ap  = (const float*)d_in[6];
  const float* bpp = (const float*)d_in[7];
  const int*   ei  = (const int*)d_in[8];
  const int*   y   = (const int*)d_in[9];
  const unsigned char* mask = (const unsigned char*)d_in[10];

  const int N  = in_sizes[0] / IN_DIM;
  const int E  = in_sizes[8] / 2;
  const int* row = ei;
  const int* col = ei + E;

  // workspace layout
  float* ws     = (float*)d_ws;
  float* z      = ws;                          // N*LAT
  float* beta   = z    + (size_t)N * LAT;      // N*NCLS
  float* buf0   = beta + (size_t)N * NCLS;     // N*NCLS
  float* buf1   = buf0 + (size_t)N * NCLS;     // N*NCLS
  float* deg    = buf1 + (size_t)N * NCLS;     // N   (becomes dinv in place)
  float* wE     = deg  + (size_t)N;            // E
  float* counts = wE   + (size_t)E;            // NCLS
  float* logpr  = counts + NCLS;               // NCLS
  __bf16* w1t   = (__bf16*)(logpr + NCLS);     // HID*IN_DIM bf16 (16B aligned)
  __bf16* w2t   = w1t + (size_t)HID * IN_DIM;  // LAT*HID bf16

  const int T = 256;
  const long long totNC = (long long)N * NCLS;
  dim3 gN((N + T - 1) / T), gE((E + T - 1) / T);
  dim3 gNC((unsigned)((totNC + T - 1) / T));

  // graph structure + prior
  init_kernel<<<gN, T, 0, stream>>>(deg, counts, N);
  count_kernel<<<gN, T, 0, stream>>>(y, mask, counts, N);
  prior_kernel<<<1, 32, 0, stream>>>(counts, logpr);
  deg_kernel<<<gE, T, 0, stream>>>(col, deg, E);
  dinv_kernel<<<gN, T, 0, stream>>>(deg, N);
  edgew_kernel<<<gE, T, 0, stream>>>(row, col, deg, wE, E);

  // one-time weight transpose/convert, then encoder (WMMA+TDM) + flow
  prep_w_kernel<<<dim3((HID * IN_DIM + T - 1) / T), T, 0, stream>>>(W1, W2, w1t, w2t);
  encoder_kernel<<<dim3((N + BM - 1) / BM), T, 0, stream>>>(x, w1t, b1, w2t, b2, z, N);
  flow_kernel<<<gNC, T, 0, stream>>>(z, z0, ap, bpp, logpr, beta, buf0, N);

  // APPNP: 10 fixed propagation steps, ping-pong buf0/buf1
  for (int k = 0; k < KSTEPS; ++k) {
    float* src = (k & 1) ? buf1 : buf0;
    float* dst = (k & 1) ? buf0 : buf1;
    teleport_kernel<<<gNC, T, 0, stream>>>(beta, dst, totNC);
    scatter_kernel<<<gE, T, 0, stream>>>(row, col, wE, src, dst, E);
  }
  // after 10 iters result is in buf0
  final_kernel<<<gNC, T, 0, stream>>>(buf0, (float*)d_out, totNC);
}
